// MaskHead_85701777424681
// MI455X (gfx1250) — matrix operand order repaired and checked by reading, not compiled, not run on hardware
//
#include <hip/hip_runtime.h>
#include <hip/hip_bf16.h>
#include <math.h>

// ---------------------------------------------------------------------------
// MaskHead on MI455X (gfx1250, wave32, WMMA 16x16x32 bf16 -> f32)
//
//   1. pack w2/w3/w4  fp32 OIHW -> bf16 [o][ky][kx][ci]   (K-contiguous)
//   2. pack wt (transposed-conv weight, flip+transpose folded in)
//   3. conv1 direct (CIN=3) fp32 NCHW -> bf16 NHWC, +bias +relu
//   4..6. conv3x3 WMMA (256->256): one block per image, image staged into
//      LDS (async global->LDS when available) zero-padded to 16x16 with a
//      264-element cell stride (132 dwords == 4 mod 64 banks -> conflict-free
//      ds_load_b128 across a half-wave). Waves process pixel tiles in groups
//      of 4: each A fragment feeds 4 WMMAs, each LDS B fragment feeds 2.
//   7. convT(stride2) WMMA by output parity class, fused relu + 1x1 conv
//      (256->3) + bias + sigmoid -> fp32 NCHW output [1024,3,28,28]
// ---------------------------------------------------------------------------

typedef __attribute__((ext_vector_type(16))) __bf16 v16bf;
typedef __attribute__((ext_vector_type(8)))  __bf16 v8bf;
typedef __attribute__((ext_vector_type(8)))  float  v8f;
typedef __attribute__((vector_size(16)))     int    v4i_t;

#define NIMG 1024
#define NPIX 196            // 14*14
#define CHN  256
#define WELEM (CHN*9*CHN)   // packed 3x3 weight elements
#define CSTRIDE 264         // padded LDS cell stride (bf16 elems); 132 dw % 64 = 4
#define SIMG_BYTES (16*16*CSTRIDE*2)    // 135168 B
#define RED_BYTES  (4*48*4)             // reduction scratch (G<=4)

#define WMMA_BF16(a, b, c) \
  __builtin_amdgcn_wmma_f32_16x16x32_bf16(false, (a), false, (b), (short)0, (c), false, false)

// Defeat LICM of weight loads across tile groups (prevents hoist+spill of
// 144 fragments to scratch). Applied to an integer offset, NOT the pointer,
// so addrspace(1) provenance survives -> global_load, not flat_load.
#define OPAQUE(x) asm volatile("" : "+v"(x))

#if __has_builtin(__builtin_amdgcn_global_load_async_to_lds_b128)
#define HAVE_ASYNC_LDS 1
#else
#define HAVE_ASYNC_LDS 0
#endif

// ---- WMMA fragment loaders (CDNA5 ISA §7.12.2) ------------------
// A (16x32 bf16, M=out-ch): lane 0-15 -> M=lane, K base 0; lane 16-31 -> K base 8.
// elems 0..7 <-> K base..base+7, elems 8..15 <-> K base+16..base+23.
__device__ __forceinline__ v16bf load_a(const __bf16* aptr) {
  const v8bf lo = *(const v8bf*)(aptr);
  const v8bf hi = *(const v8bf*)(aptr + 16);
  v16bf a;
#pragma unroll
  for (int e = 0; e < 8; ++e) { a[e] = lo[e]; a[8 + e] = hi[e]; }
  return a;
}

// B: same K order as A (elems 0..7 <-> K base..base+7, 8..15 <-> K base+16..base+23);
// 2x16B chunks (cell stride is 16B- but not 32B-aligned after bank padding).
__device__ __forceinline__ v16bf load_b(const __bf16* bptr) {
  const v8bf lo = *(const v8bf*)(bptr);
  const v8bf hi = *(const v8bf*)(bptr + 16);
  v16bf b;
#pragma unroll
  for (int e = 0; e < 8; ++e) { b[e] = lo[e]; b[8 + e] = hi[e]; }
  return b;
}

// ---- LDS staging: NHWC bf16 14x14x256 -> 16x16 cells, CSTRIDE-padded -------
__device__ __forceinline__ void stage_image(__bf16* simg,
                                            const __bf16* __restrict__ src,
                                            int tid, int nthreads) {
  v8bf z;
#pragma unroll
  for (int e = 0; e < 8; ++e) z[e] = (__bf16)0.0f;

  // 60 border cells * 32 chunks(16B) = 1920 chunks (pad tail never read)
  for (int t = tid; t < 1920; t += nthreads) {
    const int idx = t >> 5, sub = t & 31;
    int row, col;
    if (idx < 16)      { row = 0;  col = idx; }
    else if (idx < 32) { row = 15; col = idx - 16; }
    else { const int rem = idx - 32; row = 1 + (rem >> 1); col = (rem & 1) * 15; }
    *(v8bf*)(simg + ((row * 16 + col) * CSTRIDE) + sub * 8) = z;
  }
  // 196 pixels * 32 chunks(16B) = 6272 chunks
  for (int t = tid; t < 6272; t += nthreads) {
    const int p = t >> 5, sub = t & 31;
    const int y = p / 14, x = p % 14;
    __bf16* dst = simg + (((y + 1) * 16 + (x + 1)) * CSTRIDE) + sub * 8;
    const __bf16* s = src + (size_t)p * 256 + sub * 8;
#if HAVE_ASYNC_LDS
    __builtin_amdgcn_global_load_async_to_lds_b128(
        (__attribute__((address_space(1))) v4i_t*)const_cast<__bf16*>(s),
        (__attribute__((address_space(3))) v4i_t*)dst, 0, 0);
#else
    *(v8bf*)dst = *(const v8bf*)s;
#endif
  }
#if HAVE_ASYNC_LDS
#if __has_builtin(__builtin_amdgcn_s_wait_asynccnt)
  __builtin_amdgcn_s_wait_asynccnt(0);
#else
  asm volatile("s_wait_asynccnt 0x0" ::: "memory");
#endif
#endif
  __syncthreads();
}

// ---- packing kernels -------------------------------------------------------

__global__ void __launch_bounds__(256) pack_w_oihw(const float* __restrict__ w,
                                                   __bf16* __restrict__ wp) {
  int id = blockIdx.x * blockDim.x + threadIdx.x;
  if (id >= WELEM) return;
  int ci  = id & 255;
  int tap = (id >> 8) % 9;
  int o   = id / (9 * 256);
  wp[id] = (__bf16)w[((size_t)o * 256 + ci) * 9 + tap];
}

// wtp[o][ky][kx][ci] = wt[ci][o][2-ky][2-kx]
__global__ void __launch_bounds__(256) pack_wt(const float* __restrict__ wt,
                                               __bf16* __restrict__ wp) {
  int id = blockIdx.x * blockDim.x + threadIdx.x;
  if (id >= WELEM) return;
  int ci  = id & 255;
  int tap = (id >> 8) % 9;
  int o   = id / (9 * 256);
  int ky = tap / 3, kx = tap % 3;
  int ftap = (2 - ky) * 3 + (2 - kx);
  wp[id] = (__bf16)wt[((size_t)ci * 256 + o) * 9 + ftap];
}

// ---- conv1: direct 3->256, fp32 NCHW in, bf16 NHWC out, bias+relu ----------

__global__ void __launch_bounds__(256) conv1_direct(const float* __restrict__ x,
                                                    const float* __restrict__ w1,
                                                    const float* __restrict__ b1,
                                                    __bf16* __restrict__ act) {
  int id = blockIdx.x * blockDim.x + threadIdx.x;
  if (id >= NIMG * NPIX * CHN) return;
  int o  = id & 255;
  int p  = (id >> 8) % NPIX;
  int n  = id / (NPIX * 256);
  int y  = p / 14, xx = p % 14;
  float s = b1[o];
#pragma unroll
  for (int c = 0; c < 3; ++c) {
#pragma unroll
    for (int ky = 0; ky < 3; ++ky) {
      int iy = y + ky - 1;
      if ((unsigned)iy >= 14u) continue;
#pragma unroll
      for (int kx = 0; kx < 3; ++kx) {
        int ix = xx + kx - 1;
        if ((unsigned)ix >= 14u) continue;
        s += x[(((size_t)n * 3 + c) * 14 + iy) * 14 + ix] *
             w1[((o * 3 + c) * 3 + ky) * 3 + kx];
      }
    }
  }
  act[((size_t)n * NPIX + p) * CHN + o] = (__bf16)fmaxf(s, 0.0f);
}

// ---- conv3x3 tile-group worker --------------------------------------------
// Wave computes out-ch tiles {oA,oB} x G pixel tiles. Tap loop is a real loop
// (unroll 1) so only the in-flight A pair is live; ci0 and g are unrolled.

template <int G>
__device__ __forceinline__ void conv_group(const __bf16* simg,
                                           const __bf16* __restrict__ wpv,
                                           const float* __restrict__ biasv,
                                           __bf16* __restrict__ dstn,
                                           int g0, int col, int kb_b, int kb_a,
                                           int oA, int oB, int ob0, int ob1) {
  v8f c0[G], c1[G];
  int bofs[G], pidx[G];
  bool pv[G];
#pragma unroll
  for (int g = 0; g < G; ++g) {
#pragma unroll
    for (int r = 0; r < 8; ++r) { c0[g][r] = 0.f; c1[g][r] = 0.f; }
    const int p = (g0 + g) * 16 + col;
    pidx[g] = p;
    pv[g] = (p < NPIX);
    const int pc = pv[g] ? p : (NPIX - 1);    // clamp: junk columns never stored
    bofs[g] = ((pc / 14) * 16 + (pc % 14)) * CSTRIDE + kb_b;
  }
  const int aA = oA * 9, aB = oB * 9;
#pragma unroll 1
  for (int tap = 0; tap < 9; ++tap) {
    const int ky = tap / 3, kx = tap - ky * 3;
    const int toff = (ky * 16 + kx) * CSTRIDE;
    const __bf16* a0p = wpv + (size_t)(aA + tap) * CHN + kb_a;
    const __bf16* a1p = wpv + (size_t)(aB + tap) * CHN + kb_a;
#pragma unroll
    for (int ci0 = 0; ci0 < CHN; ci0 += 32) {
      const v16bf a0 = load_a(a0p + ci0);
      const v16bf a1 = load_a(a1p + ci0);
#pragma unroll
      for (int g = 0; g < G; ++g) {
        const v16bf b = load_b(simg + bofs[g] + toff + ci0);
        c0[g] = WMMA_BF16(a0, b, c0[g]);
        c1[g] = WMMA_BF16(a1, b, c1[g]);
      }
    }
  }
#pragma unroll
  for (int g = 0; g < G; ++g) {
    if (pv[g]) {
      v8bf o0, o1;
#pragma unroll
      for (int r = 0; r < 8; ++r) {
        o0[r] = (__bf16)fmaxf(c0[g][r] + biasv[ob0 + r], 0.0f);
        o1[r] = (__bf16)fmaxf(c1[g][r] + biasv[ob1 + r], 0.0f);
      }
      __bf16* dst = dstn + (size_t)pidx[g] * CHN;
      *(v8bf*)(dst + ob0) = o0;
      *(v8bf*)(dst + ob1) = o1;
    }
  }
}

// grid = 1024 images, block = 256 (8 waves); LDS-staged padded image.
__global__ void __launch_bounds__(256) conv3x3_wmma(const __bf16* __restrict__ act_in,
                                                    const __bf16* __restrict__ wp,
                                                    const float* __restrict__ bias,
                                                    __bf16* __restrict__ act_out) {
  extern __shared__ char smem[];
  __bf16* simg = (__bf16*)smem;

  const int n    = blockIdx.x;
  const int tid  = threadIdx.x;
  const int lane = tid & 31;
  const int wave = tid >> 5;

  stage_image(simg, act_in + (size_t)n * NPIX * CHN, tid, 256);

  const int col  = lane & 15;
  const int hi   = lane >> 4;
  const int kb_b = hi * 8;
  const int kb_a = hi * 8;
  const int oA  = wave * 16 + col;
  const int oB  = (wave + 8) * 16 + col;
  const int ob0 = wave * 16 + kb_a;
  const int ob1 = (wave + 8) * 16 + kb_a;

  __bf16* dstn = act_out + (size_t)n * NPIX * CHN;

  int wo = 0, bo = 0;   // opaque offsets: block LICM across groups
  OPAQUE(wo); OPAQUE(bo);
  conv_group<4>(simg, wp + wo, bias + bo, dstn, 0,  col, kb_b, kb_a, oA, oB, ob0, ob1);
  OPAQUE(wo); OPAQUE(bo);
  conv_group<4>(simg, wp + wo, bias + bo, dstn, 4,  col, kb_b, kb_a, oA, oB, ob0, ob1);
  OPAQUE(wo); OPAQUE(bo);
  conv_group<4>(simg, wp + wo, bias + bo, dstn, 8,  col, kb_b, kb_a, oA, oB, ob0, ob1);
  OPAQUE(wo); OPAQUE(bo);
  conv_group<1>(simg, wp + wo, bias + bo, dstn, 12, col, kb_b, kb_a, oA, oB, ob0, ob1);
}

// ---- convT tile-group worker -----------------------------------------------
// Output pixel (y,x)=(2j+cy, 2i+cx); parity class selects tap set:
//   cy==0: ky=1, iy=j      cy==1: ky in {0,2}, iy in {j, j+1}
// Padded row/col 15 are zero -> iy==14 / ix==14 taps are branch-free.

template <int G>
__device__ __forceinline__ void convt_group(const __bf16* simg,
                                            const __bf16* __restrict__ wtv,
                                            const float* __restrict__ btv,
                                            const float* __restrict__ w5v,
                                            const float* __restrict__ b5,
                                            float* __restrict__ outn,
                                            float* red,
                                            int g0, int cy, int cx,
                                            int tid, int col, int kb_b, int kb_a,
                                            int oA, int oB, int ob0, int ob1) {
  for (int t = tid; t < G * 48; t += 256) red[t] = 0.0f;
  __syncthreads();

  v8f c0[G], c1[G];
  int bofs[G];
  bool pv[G];
#pragma unroll
  for (int g = 0; g < G; ++g) {
#pragma unroll
    for (int r = 0; r < 8; ++r) { c0[g][r] = 0.f; c1[g][r] = 0.f; }
    const int p = (g0 + g) * 16 + col;
    pv[g] = (p < NPIX);
    const int pc = pv[g] ? p : (NPIX - 1);
    const int j = pc / 14, i = pc % 14;
    bofs[g] = ((j + 1) * 16 + (i + 1)) * CSTRIDE + kb_b;
  }

  int nky, kyA[2], dyA[2], nkx, kxA[2], dxA[2];
  if (cy == 0) { nky = 1; kyA[0] = 1; dyA[0] = 0; }
  else         { nky = 2; kyA[0] = 0; dyA[0] = 0; kyA[1] = 2; dyA[1] = 1; }
  if (cx == 0) { nkx = 1; kxA[0] = 1; dxA[0] = 0; }
  else         { nkx = 2; kxA[0] = 0; dxA[0] = 0; kxA[1] = 2; dxA[1] = 1; }

  const int aA = oA * 9, aB = oB * 9;
#pragma unroll 1
  for (int ti = 0; ti < nky; ++ti) {
#pragma unroll 1
    for (int tj = 0; tj < nkx; ++tj) {
      const int tap  = kyA[ti] * 3 + kxA[tj];
      const int toff = (dyA[ti] * 16 + dxA[tj]) * CSTRIDE;
      const __bf16* a0p = wtv + (size_t)(aA + tap) * CHN + kb_a;
      const __bf16* a1p = wtv + (size_t)(aB + tap) * CHN + kb_a;
#pragma unroll
      for (int ci0 = 0; ci0 < CHN; ci0 += 32) {
        const v16bf a0 = load_a(a0p + ci0);
        const v16bf a1 = load_a(a1p + ci0);
#pragma unroll
        for (int g = 0; g < G; ++g) {
          const v16bf b = load_b(simg + bofs[g] + toff + ci0);
          c0[g] = WMMA_BF16(a0, b, c0[g]);
          c1[g] = WMMA_BF16(a1, b, c1[g]);
        }
      }
    }
  }

  // relu(convT + bt) contracted against w5 (3x256); LDS f32 reduction
#pragma unroll
  for (int g = 0; g < G; ++g) {
    float s0 = 0.f, s1 = 0.f, s2 = 0.f;
#pragma unroll
    for (int r = 0; r < 8; ++r) {
      const int o0 = ob0 + r, o1 = ob1 + r;
      const float v0 = fmaxf(c0[g][r] + btv[o0], 0.0f);
      const float v1 = fmaxf(c1[g][r] + btv[o1], 0.0f);
      s0 += v0 * w5v[o0]       + v1 * w5v[o1];
      s1 += v0 * w5v[256 + o0] + v1 * w5v[256 + o1];
      s2 += v0 * w5v[512 + o0] + v1 * w5v[512 + o1];
    }
    if (pv[g]) {
      atomicAdd(&red[g * 48 + col * 3 + 0], s0);
      atomicAdd(&red[g * 48 + col * 3 + 1], s1);
      atomicAdd(&red[g * 48 + col * 3 + 2], s2);
    }
  }
  __syncthreads();

  for (int t = tid; t < G * 48; t += 256) {
    const int g = t / 48, r = t % 48;
    const int cl = r / 3, cc = r % 3;
    const int pq = (g0 + g) * 16 + cl;
    if (pq < NPIX) {
      const int jj = pq / 14, ii = pq % 14;
      const int yy = 2 * jj + cy, xx = 2 * ii + cx;
      const float v = red[t] + b5[cc];
      outn[(size_t)cc * 784 + yy * 28 + xx] = 1.0f / (1.0f + __expf(-v));
    }
  }
  __syncthreads();
}

// grid = 1024 images, block = 256 (8 waves, 2 out-ch tiles each = all 256).
__global__ void __launch_bounds__(256) convt_fused(const __bf16* __restrict__ act_in,
                                                   const __bf16* __restrict__ wtp,
                                                   const float* __restrict__ bt,
                                                   const float* __restrict__ w5,
                                                   const float* __restrict__ b5,
                                                   float* __restrict__ out) {
  extern __shared__ char smem[];
  __bf16* simg = (__bf16*)smem;
  float*  red  = (float*)(smem + SIMG_BYTES);

  const int n    = blockIdx.x;
  const int tid  = threadIdx.x;
  const int lane = tid & 31;
  const int wave = tid >> 5;

  stage_image(simg, act_in + (size_t)n * NPIX * CHN, tid, 256);

  const int col  = lane & 15;
  const int hi   = lane >> 4;
  const int kb_b = hi * 8;
  const int kb_a = hi * 8;
  const int oA  = wave * 16 + col;
  const int oB  = (wave + 8) * 16 + col;
  const int ob0 = wave * 16 + kb_a;
  const int ob1 = (wave + 8) * 16 + kb_a;

  float* outn = out + (size_t)n * 3 * 784;

#pragma unroll 1
  for (int cls = 0; cls < 4; ++cls) {
    const int cy = cls >> 1, cx = cls & 1;
    int wo = 0, to = 0, fo = 0;
    OPAQUE(wo); OPAQUE(to); OPAQUE(fo);
    convt_group<4>(simg, wtp + wo, bt + to, w5 + fo, b5, outn, red, 0,  cy, cx, tid, col, kb_b, kb_a, oA, oB, ob0, ob1);
    OPAQUE(wo); OPAQUE(to); OPAQUE(fo);
    convt_group<4>(simg, wtp + wo, bt + to, w5 + fo, b5, outn, red, 4,  cy, cx, tid, col, kb_b, kb_a, oA, oB, ob0, ob1);
    OPAQUE(wo); OPAQUE(to); OPAQUE(fo);
    convt_group<4>(simg, wtp + wo, bt + to, w5 + fo, b5, outn, red, 8,  cy, cx, tid, col, kb_b, kb_a, oA, oB, ob0, ob1);
    OPAQUE(wo); OPAQUE(to); OPAQUE(fo);
    convt_group<1>(simg, wtp + wo, bt + to, w5 + fo, b5, outn, red, 12, cy, cx, tid, col, kb_b, kb_a, oA, oB, ob0, ob1);
  }
}

// ---------------------------------------------------------------------------

extern "C" void kernel_launch(void* const* d_in, const int* in_sizes, int n_in,
                              void* d_out, int out_size, void* d_ws, size_t ws_size,
                              hipStream_t stream) {
  const float* x  = (const float*)d_in[0];
  const float* w1 = (const float*)d_in[1];
  const float* b1 = (const float*)d_in[2];
  const float* w2 = (const float*)d_in[3];
  const float* b2 = (const float*)d_in[4];
  const float* w3 = (const float*)d_in[5];
  const float* b3 = (const float*)d_in[6];
  const float* w4 = (const float*)d_in[7];
  const float* b4 = (const float*)d_in[8];
  const float* wt = (const float*)d_in[9];
  const float* bt = (const float*)d_in[10];
  const float* w5 = (const float*)d_in[11];
  const float* b5 = (const float*)d_in[12];

  char* ws = (char*)d_ws;
  const size_t ACT_BYTES = (size_t)NIMG * NPIX * CHN * sizeof(__bf16); // ~102.8 MB
  const size_t WP_ELEMS  = (size_t)WELEM;

  __bf16* actA = (__bf16*)ws;
  __bf16* actB = (__bf16*)(ws + ACT_BYTES);
  __bf16* wp2  = (__bf16*)(ws + 2 * ACT_BYTES);
  __bf16* wp3  = wp2 + WP_ELEMS;
  __bf16* wp4  = wp3 + WP_ELEMS;
  __bf16* wtp  = wp4 + WP_ELEMS;

  const int wgrid = (WELEM + 255) / 256;
  pack_w_oihw<<<wgrid, 256, 0, stream>>>(w2, wp2);
  pack_w_oihw<<<wgrid, 256, 0, stream>>>(w3, wp3);
  pack_w_oihw<<<wgrid, 256, 0, stream>>>(w4, wp4);
  pack_wt    <<<wgrid, 256, 0, stream>>>(wt, wtp);

  const int t1 = NIMG * NPIX * CHN;
  conv1_direct<<<(t1 + 255) / 256, 256, 0, stream>>>(x, w1, b1, actA);

  conv3x3_wmma<<<NIMG, 256, SIMG_BYTES, stream>>>(actA, wp2, b2, actB);
  conv3x3_wmma<<<NIMG, 256, SIMG_BYTES, stream>>>(actB, wp3, b3, actA);
  conv3x3_wmma<<<NIMG, 256, SIMG_BYTES, stream>>>(actA, wp4, b4, actB);

  convt_fused<<<NIMG, 256, SIMG_BYTES + RED_BYTES, stream>>>(actB, wtp, bt, w5, b5,
                                                             (float*)d_out);
}
